// Attention_62010737820049
// MI455X (gfx1250) — matrix operand order repaired and checked by reading, not compiled, not run on hardware
//
#include <hip/hip_runtime.h>
#include <stdint.h>

typedef __attribute__((ext_vector_type(16))) __bf16 v16bf;
typedef __attribute__((ext_vector_type(8)))  float  v8f;

// ---------- helpers ----------
__device__ __forceinline__ unsigned short f32_to_bf16_rne(float f) {
  union { float f; uint32_t u; } x; x.f = f;
  uint32_t u = x.u;
  u += 0x7FFFu + ((u >> 16) & 1u);     // round-to-nearest-even
  return (unsigned short)(u >> 16);
}

union Frag {
  v16bf v;
  uint4 q2[2];
};

// ---------- elementwise conversion ----------
__global__ void k_f32_to_bf16(const float* __restrict__ src,
                              unsigned short* __restrict__ dst, long long n) {
  long long i = (long long)blockIdx.x * blockDim.x + threadIdx.x;
  long long stride = (long long)gridDim.x * blockDim.x;
  for (; i < n; i += stride) dst[i] = f32_to_bf16_rne(src[i]);
}

// hf = concat(h, x) along sequence dim, converted to bf16
__global__ void k_pack_hf(const float* __restrict__ h, const float* __restrict__ x,
                          unsigned short* __restrict__ hf,
                          int B, int M, int C, int D) {
  const int W = M + C;
  long long n = (long long)B * W * D;
  long long i = (long long)blockIdx.x * blockDim.x + threadIdx.x;
  long long stride = (long long)gridDim.x * blockDim.x;
  for (; i < n; i += stride) {
    int d = (int)(i % D);
    long long r = i / D;
    int w = (int)(r % W);
    int b = (int)(r / W);
    float v = (w < M) ? h[((long long)b * M + w) * D + d]
                      : x[((long long)b * C + (w - M)) * D + d];
    hf[i] = f32_to_bf16_rne(v);
  }
}

// ---------- batched NT GEMM (LDS double-buffered): ----------
// Out[m,n] = sum_k A[m,k]*B[n,k] (+bias)
// Block tile 128x128, 8 waves, wave tile 32x64 (2x4 WMMA 16x16 tiles).
// modes: 0=f32 out, 1=bf16 out, 2=bf16+relu, 3=bf16 out + bf16 (acc+u) out2,
//        4=bf16 transposed store Out[n*ldo + m]
#define LDS_PITCH 40   // 32 K-elements + 8 pad (80B row pitch, conflict-free)
__global__ __launch_bounds__(256)
void k_gemm_bf16_nt(const unsigned short* __restrict__ A, long long sAb, int lda,
                    const unsigned short* __restrict__ Bm, long long sBb, int ldb,
                    float* __restrict__ OutF, unsigned short* __restrict__ OutH,
                    unsigned short* __restrict__ Out2H,
                    long long sOb, int ldo,
                    const float* __restrict__ bias,
                    const float* __restrict__ uadd, int ldu,
                    int K, int mode) {
  __shared__ __align__(16) unsigned short As[2][128 * LDS_PITCH];
  __shared__ __align__(16) unsigned short Bs[2][128 * LDS_PITCH];

  const int z = blockIdx.z;
  A  += (long long)z * sAb;
  Bm += (long long)z * sBb;
  const long long obase = (long long)z * sOb;

  const int t    = threadIdx.x;
  const int lane = t & 31;
  const int wave = t >> 5;
  const int wm = wave & 3;              // 4 waves along M (32 rows each)
  const int wn = wave >> 2;             // 2 waves along N (64 cols each)
  const int blockRow = blockIdx.x * 128;
  const int blockCol = blockIdx.y * 128;
  const int m = lane & 15;
  const int g = lane >> 4;

  // cooperative staging: thread t copies 32B of row (t>>1), elem off (t&1)*16
  const int srow = t >> 1;
  const int soff = (t & 1) * 16;
  const unsigned short* gA = A  + (long long)(blockRow + srow) * lda + soff;
  const unsigned short* gB = Bm + (long long)(blockCol + srow) * ldb + soff;
  const int sidx = srow * LDS_PITCH + soff;

  v8f acc[2][4] = {};
  uint4 ra0, ra1, rb0, rb1;

  auto gload = [&](int k0) {
    const unsigned short* pa = gA + k0;
    const unsigned short* pb = gB + k0;
    ra0 = *reinterpret_cast<const uint4*>(pa);
    ra1 = *reinterpret_cast<const uint4*>(pa + 8);
    rb0 = *reinterpret_cast<const uint4*>(pb);
    rb1 = *reinterpret_cast<const uint4*>(pb + 8);
  };
  auto sstore = [&](int buf) {
    *reinterpret_cast<uint4*>(&As[buf][sidx])     = ra0;
    *reinterpret_cast<uint4*>(&As[buf][sidx + 8]) = ra1;
    *reinterpret_cast<uint4*>(&Bs[buf][sidx])     = rb0;
    *reinterpret_cast<uint4*>(&Bs[buf][sidx + 8]) = rb1;
  };
  auto compute = [&](int buf) {
    Frag fa[2], fb[4];
#pragma unroll
    for (int tm = 0; tm < 2; ++tm) {
      // A 16x32 frag: lane row m, K halves {g*8.., 16+g*8..} (ISA 7.12.2)
      const unsigned short* p = &As[buf][(wm * 32 + tm * 16 + m) * LDS_PITCH + g * 8];
      fa[tm].q2[0] = *reinterpret_cast<const uint4*>(p);
      fa[tm].q2[1] = *reinterpret_cast<const uint4*>(p + 16);
    }
#pragma unroll
    for (int tn = 0; tn < 4; ++tn) {
      // B 32x16 frag: lane col m, K halves {g*8.., 16+g*8..}
      const unsigned short* p = &Bs[buf][(wn * 64 + tn * 16 + m) * LDS_PITCH + g * 8];
      fb[tn].q2[0] = *reinterpret_cast<const uint4*>(p);
      fb[tn].q2[1] = *reinterpret_cast<const uint4*>(p + 16);
    }
#pragma unroll
    for (int tm = 0; tm < 2; ++tm)
#pragma unroll
      for (int tn = 0; tn < 4; ++tn)
        acc[tm][tn] = __builtin_amdgcn_wmma_f32_16x16x32_bf16(
            false, fa[tm].v, false, fb[tn].v, (short)0, acc[tm][tn], false, false);
  };

  gload(0);
  sstore(0);
  __syncthreads();
  int cur = 0;
  for (int k0 = 32; k0 < K; k0 += 32) {
    gload(k0);              // next slab global->regs (overlaps compute)
    compute(cur);           // WMMAs from current LDS slab
    sstore(cur ^ 1);        // regs->LDS (other buffer)
    __syncthreads();
    cur ^= 1;
  }
  compute(cur);

  // ---- epilogue ----
#pragma unroll
  for (int tm = 0; tm < 2; ++tm) {
#pragma unroll
    for (int tn = 0; tn < 4; ++tn) {
      const int r0 = blockRow + wm * 32 + tm * 16 + g * 8; // lanes>=16 hold M+8
      const int c0 = blockCol + wn * 64 + tn * 16 + m;
      const float bv = bias ? bias[c0] : 0.0f;
#pragma unroll
      for (int i = 0; i < 8; ++i) {
        const int rr = r0 + i;
        float v = acc[tm][tn][i] + bv;
        if (mode == 0) {
          OutF[obase + (long long)rr * ldo + c0] = v;
        } else if (mode == 1) {
          OutH[obase + (long long)rr * ldo + c0] = f32_to_bf16_rne(v);
        } else if (mode == 2) {
          OutH[obase + (long long)rr * ldo + c0] = f32_to_bf16_rne(fmaxf(v, 0.0f));
        } else if (mode == 3) {
          const long long idx = obase + (long long)rr * ldo + c0;
          OutH[idx]  = f32_to_bf16_rne(v);
          Out2H[idx] = f32_to_bf16_rne(v + uadd[(long long)rr * ldu + c0]);
        } else { // mode 4: transposed store
          OutH[obase + (long long)c0 * ldo + rr] = f32_to_bf16_rne(v);
        }
      }
    }
  }
}

// ---------- fused score combine + circulant shift + mask + softmax ----------
// s[w] = mask(w<=c+M) ? (qk[w] + braw[(w-c+C-1) mod W] + s2[w]) * scale : 0
__global__ __launch_bounds__(256)
void k_softmax(const float* __restrict__ qk, const float* __restrict__ braw,
               const float* __restrict__ s2, unsigned short* __restrict__ att,
               int C, int W, int Mlen, float scale) {
  const int c = blockIdx.x, b = blockIdx.y;
  const long long row = (long long)b * C + c;
  const float* qkr = qk   + row * W;
  const float* brr = braw + row * W;
  const float* s2r = s2   + (long long)c * W;
  unsigned short* ar = att + row * W;

  __shared__ float red[256];
  const int t = threadIdx.x;
  float vals[4];
  float lmax = -3.4e38f;
#pragma unroll
  for (int j = 0; j < 4; ++j) {
    const int w = t + j * 256;
    int sh = w - c + (C - 1);
    if (sh >= W) sh -= W;
    float s = qkr[w] + brr[sh] + s2r[w];
    s = (w <= c + Mlen) ? s * scale : 0.0f;   // reference masks to 0 (not -inf)
    vals[j] = s;
    lmax = fmaxf(lmax, s);
  }
  red[t] = lmax; __syncthreads();
  for (int off = 128; off > 0; off >>= 1) {
    if (t < off) red[t] = fmaxf(red[t], red[t + off]);
    __syncthreads();
  }
  const float mx = red[0]; __syncthreads();

  float lsum = 0.0f;
#pragma unroll
  for (int j = 0; j < 4; ++j) { vals[j] = __expf(vals[j] - mx); lsum += vals[j]; }
  red[t] = lsum; __syncthreads();
  for (int off = 128; off > 0; off >>= 1) {
    if (t < off) red[t] += red[t + off];
    __syncthreads();
  }
  const float inv = 1.0f / red[0];
#pragma unroll
  for (int j = 0; j < 4; ++j) ar[t + j * 256] = f32_to_bf16_rne(vals[j] * inv);
}

// ---------- LayerNorm over last dim (D=1024), bf16 out ----------
__global__ __launch_bounds__(256)
void k_layernorm(const float* __restrict__ o, const float* __restrict__ gamma,
                 const float* __restrict__ beta, unsigned short* __restrict__ out,
                 int D, float eps) {
  const long long row = blockIdx.x;
  const float* orow = o + row * D;
  unsigned short* orow_o = out + row * D;
  __shared__ float rs[256], rq[256];
  const int t = threadIdx.x;
  float xv[4], s1 = 0.0f, s2 = 0.0f;
#pragma unroll
  for (int j = 0; j < 4; ++j) {
    xv[j] = orow[t + j * 256];
    s1 += xv[j];
    s2 += xv[j] * xv[j];
  }
  rs[t] = s1; rq[t] = s2; __syncthreads();
  for (int off = 128; off > 0; off >>= 1) {
    if (t < off) { rs[t] += rs[t + off]; rq[t] += rq[t + off]; }
    __syncthreads();
  }
  const float mu  = rs[0] / (float)D;
  const float var = rq[0] / (float)D - mu * mu;
  const float rstd = rsqrtf(var + eps);
#pragma unroll
  for (int j = 0; j < 4; ++j) {
    const int d = t + j * 256;
    orow_o[d] = f32_to_bf16_rne((xv[j] - mu) * rstd * gamma[d] + beta[d]);
  }
}

// ---------- host launcher ----------
extern "C" void kernel_launch(void* const* d_in, const int* in_sizes, int n_in,
                              void* d_out, int out_size, void* d_ws, size_t ws_size,
                              hipStream_t stream) {
  (void)in_sizes; (void)n_in; (void)out_size; (void)ws_size;
  const int B = 32, C = 512, M = 512, D = 1024, W = 1024, D4 = 4096;

  const float* x    = (const float*)d_in[0];
  const float* h    = (const float*)d_in[1];
  const float* Wq   = (const float*)d_in[2];
  const float* bq   = (const float*)d_in[3];
  const float* Wke  = (const float*)d_in[4];
  const float* bke  = (const float*)d_in[5];
  const float* Wkr  = (const float*)d_in[6];
  const float* bkr  = (const float*)d_in[7];
  const float* Wv   = (const float*)d_in[8];
  const float* bv   = (const float*)d_in[9];
  const float* Wmlp = (const float*)d_in[10];
  const float* bmlp = (const float*)d_in[11];
  const float* gamma= (const float*)d_in[12];
  const float* beta = (const float*)d_in[13];
  const float* Wf1  = (const float*)d_in[14];
  const float* bf1  = (const float*)d_in[15];
  const float* Wf2  = (const float*)d_in[16];
  const float* bf2  = (const float*)d_in[17];
  const float* u    = (const float*)d_in[18];
  const float* v    = (const float*)d_in[19];
  const float* r    = (const float*)d_in[20];
  float* out        = (float*)d_out;

  // ---- workspace carve ----
  size_t off = 0;
  auto carve = [&](size_t bytes) -> char* {
    char* p = (char*)d_ws + off;
    off += (bytes + 255) & ~(size_t)255;
    return p;
  };
  typedef unsigned short u16;
  u16* hf_b   = (u16*)carve((size_t)B * W * D * 2);
  u16* Wq_b   = (u16*)carve((size_t)D * D * 2);
  u16* Wke_b  = (u16*)carve((size_t)D * D * 2);
  u16* Wkr_b  = (u16*)carve((size_t)D * D * 2);
  u16* Wv_b   = (u16*)carve((size_t)D * D * 2);
  u16* Wmlp_b = (u16*)carve((size_t)D * D * 2);
  u16* Wf1_b  = (u16*)carve((size_t)D4 * D * 2);
  u16* Wf2_b  = (u16*)carve((size_t)D * D4 * 2);
  u16* r_b    = (u16*)carve((size_t)W * D * 2);
  u16* v_b    = (u16*)carve((size_t)C * D * 2);
  u16* q_b    = (u16*)carve((size_t)B * C * D * 2);
  u16* qu_b   = (u16*)carve((size_t)B * C * D * 2);
  u16* k_b    = (u16*)carve((size_t)B * W * D * 2);
  u16* valT_b = (u16*)carve((size_t)B * D * W * 2);
  u16* kr_b   = (u16*)carve((size_t)W * D * 2);
  float* S2   = (float*)carve((size_t)C * W * 4);
  float* qk_raw = (float*)carve((size_t)B * C * W * 4);
  float* b_raw  = (float*)carve((size_t)B * C * W * 4);
  u16* att_b  = (u16*)carve((size_t)B * C * W * 2);
  u16* o_b    = (u16*)carve((size_t)B * C * D * 2);
  float* o_mlp= (float*)carve((size_t)B * C * D * 4);
  u16* ln_b   = (u16*)carve((size_t)B * C * D * 2);
  // FFN hidden activation reuses the (dead by then) score buffers: 128MB
  u16* f1_b   = (u16*)qk_raw;

  auto cvt = [&](const float* s, u16* d, long long n) {
    int blocks = (int)((n + 255) / 256);
    k_f32_to_bf16<<<blocks, 256, 0, stream>>>(s, d, n);
  };
  auto gemm = [&](const u16* A, long long sAb, int lda,
                  const u16* Bm, long long sBb, int ldb,
                  float* OF, u16* OH, u16* O2, long long sOb, int ldo,
                  const float* bias, const float* uadd, int ldu,
                  int Md, int Nd, int Kd, int batch, int mode) {
    dim3 g(Md / 128, Nd / 128, batch);
    k_gemm_bf16_nt<<<g, 256, 0, stream>>>(A, sAb, lda, Bm, sBb, ldb,
                                          OF, OH, O2, sOb, ldo,
                                          bias, uadd, ldu, Kd, mode);
  };

  // ---- phase 0: pack + convert ----
  {
    long long n = (long long)B * W * D;
    k_pack_hf<<<(int)((n + 255) / 256), 256, 0, stream>>>(h, x, hf_b, B, M, C, D);
  }
  cvt(Wq,  Wq_b,  (long long)D * D);
  cvt(Wke, Wke_b, (long long)D * D);
  cvt(Wkr, Wkr_b, (long long)D * D);
  cvt(Wv,  Wv_b,  (long long)D * D);
  cvt(Wmlp,Wmlp_b,(long long)D * D);
  cvt(Wf1, Wf1_b, (long long)D4 * D);
  cvt(Wf2, Wf2_b, (long long)D * D4);
  cvt(r,   r_b,   (long long)W * D);
  cvt(v,   v_b,   (long long)C * D);

  // ---- phase 1: projections ----
  // q (A = x rows of hf) -> q_b and qu_b = bf16(q + u)
  gemm(hf_b + (long long)M * D, (long long)W * D, D, Wq_b, 0, D,
       nullptr, q_b, qu_b, (long long)C * D, D, bq, u, D,
       C, D, D, B, /*mode=*/3);
  // k
  gemm(hf_b, (long long)W * D, D, Wke_b, 0, D,
       nullptr, k_b, nullptr, (long long)W * D, D, bke, nullptr, 0,
       W, D, D, B, /*mode=*/1);
  // val, stored transposed (D x W per batch)
  gemm(hf_b, (long long)W * D, D, Wv_b, 0, D,
       nullptr, valT_b, nullptr, (long long)D * W, W, bv, nullptr, 0,
       W, D, D, B, /*mode=*/4);
  // kr = r @ Wkr^T
  gemm(r_b, 0, D, Wkr_b, 0, D,
       nullptr, kr_b, nullptr, 0, D, bkr, nullptr, 0,
       W, D, D, 1, /*mode=*/1);
  // S2 = v @ kr^T (f32)
  gemm(v_b, 0, D, kr_b, 0, D,
       S2, nullptr, nullptr, 0, W, nullptr, nullptr, 0,
       C, W, D, 1, /*mode=*/0);

  // ---- phase 2: attention scores ----
  // qk_raw = (q+u) @ k^T
  gemm(qu_b, (long long)C * D, D, k_b, (long long)W * D, D,
       qk_raw, nullptr, nullptr, (long long)C * W, W, nullptr, nullptr, 0,
       C, W, D, B, /*mode=*/0);
  // b_raw = q @ kr^T
  gemm(q_b, (long long)C * D, D, kr_b, 0, D,
       b_raw, nullptr, nullptr, (long long)C * W, W, nullptr, nullptr, 0,
       C, W, D, B, /*mode=*/0);
  // fused shift + mask + softmax -> att_b (bf16)
  {
    dim3 g(C, B);
    k_softmax<<<g, 256, 0, stream>>>(qk_raw, b_raw, S2, att_b, C, W, M,
                                     1.0f / 32.0f /* 1/sqrt(1024) */);
  }

  // ---- phase 3: attention output + MLP + LN ----
  // o = att @ val  (NT against valT)
  gemm(att_b, (long long)C * W, W, valT_b, (long long)D * W, W,
       nullptr, o_b, nullptr, (long long)C * D, D, nullptr, nullptr, 0,
       C, D, W, B, /*mode=*/1);
  // o_mlp = o @ Wmlp^T + bmlp (f32)
  gemm(o_b, (long long)C * D, D, Wmlp_b, 0, D,
       o_mlp, nullptr, nullptr, (long long)C * D, D, bmlp, nullptr, 0,
       C, D, D, B, /*mode=*/0);
  // layernorm -> ln_b (bf16)
  k_layernorm<<<B * C, 256, 0, stream>>>(o_mlp, gamma, beta, ln_b, D, 1e-5f);

  // ---- phase 4: FFN ----
  // f1 = relu(ln @ Wf1^T + bf1) (bf16), flattened M = B*C
  gemm(ln_b, 0, D, Wf1_b, 0, D,
       nullptr, f1_b, nullptr, 0, D4, bf1, nullptr, 0,
       B * C, D4, D, 1, /*mode=*/2);
  // out = f1 @ Wf2^T + bf2 (f32) -> d_out
  gemm(f1_b, 0, D4, Wf2_b, 0, D4,
       out, nullptr, nullptr, 0, D, bf2, nullptr, 0,
       B * C, D, D4, 1, /*mode=*/0);
}
